// SoftEmbedding_42494406427453
// MI455X (gfx1250) — matrix operand order repaired and checked by reading, not compiled, not run on hardware
//
#include <hip/hip_runtime.h>

typedef __attribute__((ext_vector_type(16))) _Float16 v16h;
typedef __attribute__((ext_vector_type(8)))  float    v8f;

#define NUM_EMB   512
#define EMB_DIM   64
#define KPITCH    520   // f16 per embT row: 512 + 8 pad -> dword stride 260 (== 4 mod 64), conflict-free b128 B-frag loads
#define TOK_PER_BLOCK 128
#define LOG2E     1.4426950408889634f

union BFrag { uint4 q[2]; v16h v; };

__global__ __launch_bounds__(256) void soft_embedding_kernel(
    const float* __restrict__ x_in,
    const float* __restrict__ proj_w,
    const float* __restrict__ proj_b,
    const float* __restrict__ emb,
    float* __restrict__ out)
{
    extern __shared__ char smem_raw[];
    _Float16* embT  = (_Float16*)smem_raw;                        // [EMB_DIM][KPITCH] f16, transposed emb
    float*    sw    = (float*)(smem_raw + EMB_DIM * KPITCH * 2);  // [NUM_EMB] w * log2(e)
    float*    sb    = sw + NUM_EMB;                               // [NUM_EMB] b * log2(e)
    float*    red   = sb + NUM_EMB;                               // [8][3] per-wave {max w, max -w, max b}
    float*    ssums = red + 24;                                   // [8][16] per-token softmax denominators

    const int tid  = threadIdx.x;
    const int lane = tid & 31;
    const int wv   = tid >> 5;

    // ---- Stage w/b (pre-scaled by log2 e) and gather per-thread extrema for the shift bound ----
    float lw = -1e30f, lnw = -1e30f, lb = -1e30f;
    for (int i = tid; i < NUM_EMB; i += 256) {
        float wval = proj_w[i] * LOG2E;
        float bval = proj_b[i] * LOG2E;
        sw[i] = wval;
        sb[i] = bval;
        lw  = fmaxf(lw,  wval);
        lnw = fmaxf(lnw, -wval);
        lb  = fmaxf(lb,  bval);
    }
#pragma unroll
    for (int off = 16; off > 0; off >>= 1) {       // wave32 all-reduce (max)
        lw  = fmaxf(lw,  __shfl_xor(lw,  off));
        lnw = fmaxf(lnw, __shfl_xor(lnw, off));
        lb  = fmaxf(lb,  __shfl_xor(lb,  off));
    }
    if (lane == 0) { red[wv * 3] = lw; red[wv * 3 + 1] = lnw; red[wv * 3 + 2] = lb; }

    // ---- Stage emb_table transposed into LDS as f16: embT[n][k] = emb[k][n] ----
    const float4* emb4 = (const float4*)emb;
    for (int i = tid; i < NUM_EMB * EMB_DIM / 4; i += 256) {
        float4 v = emb4[i];
        int k = i >> 4;
        int n = (i & 15) << 2;
        _Float16* dst = &embT[(size_t)n * KPITCH + k];
        dst[0]          = (_Float16)v.x;
        dst[KPITCH]     = (_Float16)v.y;
        dst[2 * KPITCH] = (_Float16)v.z;
        dst[3 * KPITCH] = (_Float16)v.w;
    }
    __syncthreads();

    float Mw = red[0], Mnw = red[1], Mb = red[2];
#pragma unroll
    for (int j = 1; j < 8; ++j) {
        Mw  = fmaxf(Mw,  red[j * 3]);
        Mnw = fmaxf(Mnw, red[j * 3 + 1]);
        Mb  = fmaxf(Mb,  red[j * 3 + 2]);
    }

    const int  tokl = lane & 15;       // token-in-tile = A-row M = B-column N
    const int  half = lane >> 4;
    const long tile = (long)blockIdx.x * TOK_PER_BLOCK + wv * 16;

    const float x = x_in[tile + tokl];
    // Upper bound of this token's max exponent (log2 units): all weights 2^(t-shift) <= 1, f16-safe
    const float shift = fmaxf(x * Mw, -x * Mnw) + Mb;

    // ---- Single fused sweep: exp -> denominator (f32) + A fragment (f16) -> 4 WMMAs per K-step ----
    v8f acc[4] = {};
    float ps0 = 0.f, ps1 = 0.f, ps2 = 0.f, ps3 = 0.f;

    for (int kt = 0; kt < 16; ++kt) {
        const int kb = kt * 32;

        // Prefetch the 4 B fragments (dense 32x16 f16: lanes 0-15 K 0..7/16..23, lanes 16-31 K 8..15/24..31)
        BFrag bf[4];
        const _Float16* bbase = &embT[(size_t)tokl * KPITCH + kb + half * 8];
#pragma unroll
        for (int nt = 0; nt < 4; ++nt) {
            const uint4* p = (const uint4*)(bbase + (size_t)nt * 16 * KPITCH);
            bf[nt].q[0] = p[0];
            bf[nt].q[1] = p[2];
        }

        // Prefetch w/b quads covering this lane's 16 A-layout K slots
        const float4* wq = (const float4*)(sw + kb + half * 8);
        const float4* bq = (const float4*)(sb + kb + half * 8);
        float4 w0 = wq[0], w1 = wq[1], w2 = wq[4], w3 = wq[5];
        float4 b0 = bq[0], b1 = bq[1], b2 = bq[4], b3 = bq[5];

        // e = 2^(x*w + b - shift); accumulate denominator in f32, pack A in f16
        float e0  = __builtin_amdgcn_exp2f(fmaf(x, w0.x, b0.x) - shift);
        float e1  = __builtin_amdgcn_exp2f(fmaf(x, w0.y, b0.y) - shift);
        float e2  = __builtin_amdgcn_exp2f(fmaf(x, w0.z, b0.z) - shift);
        float e3  = __builtin_amdgcn_exp2f(fmaf(x, w0.w, b0.w) - shift);
        float e4  = __builtin_amdgcn_exp2f(fmaf(x, w1.x, b1.x) - shift);
        float e5  = __builtin_amdgcn_exp2f(fmaf(x, w1.y, b1.y) - shift);
        float e6  = __builtin_amdgcn_exp2f(fmaf(x, w1.z, b1.z) - shift);
        float e7  = __builtin_amdgcn_exp2f(fmaf(x, w1.w, b1.w) - shift);
        float e8  = __builtin_amdgcn_exp2f(fmaf(x, w2.x, b2.x) - shift);
        float e9  = __builtin_amdgcn_exp2f(fmaf(x, w2.y, b2.y) - shift);
        float e10 = __builtin_amdgcn_exp2f(fmaf(x, w2.z, b2.z) - shift);
        float e11 = __builtin_amdgcn_exp2f(fmaf(x, w2.w, b2.w) - shift);
        float e12 = __builtin_amdgcn_exp2f(fmaf(x, w3.x, b3.x) - shift);
        float e13 = __builtin_amdgcn_exp2f(fmaf(x, w3.y, b3.y) - shift);
        float e14 = __builtin_amdgcn_exp2f(fmaf(x, w3.z, b3.z) - shift);
        float e15 = __builtin_amdgcn_exp2f(fmaf(x, w3.w, b3.w) - shift);

        ps0 += (e0 + e1) + (e2  + e3);
        ps1 += (e4 + e5) + (e6  + e7);
        ps2 += (e8 + e9) + (e10 + e11);
        ps3 += (e12 + e13) + (e14 + e15);

        // A fragment (16-bit 16x32 layout; half 0: K kb+0..7 / kb+16..23, half 1: +8)
        v16h a;
        a[0]  = (_Float16)e0;   a[1]  = (_Float16)e1;
        a[2]  = (_Float16)e2;   a[3]  = (_Float16)e3;
        a[4]  = (_Float16)e4;   a[5]  = (_Float16)e5;
        a[6]  = (_Float16)e6;   a[7]  = (_Float16)e7;
        a[8]  = (_Float16)e8;   a[9]  = (_Float16)e9;
        a[10] = (_Float16)e10;  a[11] = (_Float16)e11;
        a[12] = (_Float16)e12;  a[13] = (_Float16)e13;
        a[14] = (_Float16)e14;  a[15] = (_Float16)e15;

#pragma unroll
        for (int nt = 0; nt < 4; ++nt) {
            acc[nt] = __builtin_amdgcn_wmma_f32_16x16x32_f16(
                false, a, false, bf[nt].v, (short)0, acc[nt], false, false);
        }
    }

    // ---- Per-token denominator -> LDS; each lane then picks up the 8 sums for its C rows ----
    float ssum = (ps0 + ps1) + (ps2 + ps3);
    ssum += __shfl_xor(ssum, 16);                 // combine the two K-halves per token
    ssums[wv * 16 + tokl] = ssum;                 // both halves write identical value (benign)
    asm volatile("s_wait_dscnt 0x0" ::: "memory"); // wave-internal DS ordering; no block barrier needed

    const float4* sv = (const float4*)(ssums + wv * 16 + half * 8);
    float4 sA = sv[0], sB = sv[1];
    float inv[8];
    inv[0] = __builtin_amdgcn_rcpf(sA.x);
    inv[1] = __builtin_amdgcn_rcpf(sA.y);
    inv[2] = __builtin_amdgcn_rcpf(sA.z);
    inv[3] = __builtin_amdgcn_rcpf(sA.w);
    inv[4] = __builtin_amdgcn_rcpf(sB.x);
    inv[5] = __builtin_amdgcn_rcpf(sB.y);
    inv[6] = __builtin_amdgcn_rcpf(sB.z);
    inv[7] = __builtin_amdgcn_rcpf(sB.w);

    // ---- Writeback: lanes 0-15 hold M 0..7 (VGPR r), lanes 16-31 hold M 8..15; N = lane&15 ----
    const long rowbase = tile + half * 8;
#pragma unroll
    for (int nt = 0; nt < 4; ++nt) {
#pragma unroll
        for (int r = 0; r < 8; ++r) {
            out[(rowbase + r) * EMB_DIM + nt * 16 + tokl] = acc[nt][r] * inv[r];
        }
    }
}

extern "C" void kernel_launch(void* const* d_in, const int* in_sizes, int n_in,
                              void* d_out, int out_size, void* d_ws, size_t ws_size,
                              hipStream_t stream) {
    const float* x    = (const float*)d_in[0];   // (B,S,1) f32, T = B*S elements
    const float* pw   = (const float*)d_in[1];   // (512,1)
    const float* pb   = (const float*)d_in[2];   // (512,)
    const float* embp = (const float*)d_in[3];   // (512,64)
    float*       outp = (float*)d_out;           // (B,S,64)

    const int T = in_sizes[0];                   // 204800, exactly divisible by 128
    const int blocks = T / TOK_PER_BLOCK;
    const size_t shmem = (size_t)EMB_DIM * KPITCH * 2 + 2 * (size_t)NUM_EMB * 4
                       + 24 * 4 + 128 * 4;       // embT + sw/sb + red + ssums (~71.3 KB)

    soft_embedding_kernel<<<blocks, 256, shmem, stream>>>(x, pw, pb, embp, outp);
}
